// MultiCenterLoss_56521769615882
// MI455X (gfx1250) — matrix operand order repaired and checked by reading, not compiled, not run on hardware
//
#include <hip/hip_runtime.h>
#include <hip/hip_bf16.h>

// MultiCenterLoss for MI455X (gfx1250, wave32).
//
// Pass 1: per-wave 16-sample tile. Squared distance = diagonal of M x M^T,
//   computed with v_wmma_f32_16x16x32_f16 (f32 accumulate); M = (x - c)
//   diffs built in f32 and narrowed to f16 (v_cvt_pk_f16_f32).
//   Lane pairs (l, l+16) share a sample row: lo lanes load features 0..15
//   of each 32-chunk, hi lanes 16..31 (exact 1.0x HBM traffic for x), and
//   the A operand reuses each lane's own diffs, like B (swizzle results unused).
// Pass 2: deterministic single-block tree reduction -> mean.
// d_ws needs 32768 * 4 = 128 KB.

#define BATCH 32768
#define FEAT 512
#define KCEN 2

typedef __attribute__((ext_vector_type(16))) _Float16 v16h;
typedef __attribute__((ext_vector_type(2)))  _Float16 v2h;
typedef __attribute__((ext_vector_type(8)))  float    v8f;

union V16H { v16h v; unsigned int u[8]; };

#define SWAPX16 0x401F   // ds_swizzle: group-of-32, xor_mask=0x10, and_mask=0x1f

__device__ __forceinline__ float4 f4sub(float4 a, float4 b) {
    return make_float4(a.x - b.x, a.y - b.y, a.z - b.z, a.w - b.w);
}

// Two f32 -> packed 2 x f16 in one dword (lowers to v_cvt_pk_f16_f32).
__device__ __forceinline__ unsigned int pk2(float a, float b) {
    v2h h; h[0] = (_Float16)a; h[1] = (_Float16)b;
    return __builtin_bit_cast(unsigned int, h);
}

// Extract v[i] for runtime i in [0,8) without scratch (7 v_cndmask).
__device__ __forceinline__ float pick8(v8f v, int i) {
    float r = v[0];
#pragma unroll
    for (int j = 1; j < 8; ++j) r = (i == j) ? (float)v[j] : r;
    return r;
}

__global__ __launch_bounds__(256)
void mcl_dist_kernel(const float* __restrict__ x,
                     const int* __restrict__ labels,
                     const float* __restrict__ centers,
                     float* __restrict__ ws) {
    const int lane  = threadIdx.x & 31;
    const int wave  = threadIdx.x >> 5;
    const int sbase = blockIdx.x * 128 + wave * 16;   // 8 waves x 16 samples
    const int row   = lane & 15;                      // sample row within tile
    const int s     = sbase + row;
    const bool hi   = lane >= 16;

    const int lbl = labels[s];
    const float* xrow  = x + (size_t)s * FEAT;
    const float* c0row = centers + ((size_t)lbl * KCEN + 0) * FEAT;
    const float* c1row = centers + ((size_t)lbl * KCEN + 1) * FEAT;

    // lo lanes own features [0,16), hi lanes own [16,32) of each chunk.
    const int fb = hi ? 16 : 0;

    v8f acc0 = {};   // running sum of squared diffs vs center 0 (diag of M M^T)
    v8f acc1 = {};   // vs center 1

    for (int q = 0; q < 16; ++q) {
        const float4* xp = (const float4*)(xrow + q * 32 + fb);
        const float4 x0 = xp[0], x1 = xp[1], x2 = xp[2], x3 = xp[3];

#pragma unroll
        for (int k = 0; k < 2; ++k) {
            const float* crow = k ? c1row : c0row;
            const float4* cp = (const float4*)(crow + q * 32 + fb);
            const float4 d0 = f4sub(x0, cp[0]);
            const float4 d1 = f4sub(x1, cp[1]);
            const float4 d2 = f4sub(x2, cp[2]);
            const float4 d3 = f4sub(x3, cp[3]);

            // w[i] = f16 diff pair for local features (2i, 2i+1) + fb
            const unsigned int w0 = pk2(d0.x, d0.y), w1 = pk2(d0.z, d0.w);
            const unsigned int w2 = pk2(d1.x, d1.y), w3 = pk2(d1.z, d1.w);
            const unsigned int w4 = pk2(d2.x, d2.y), w5 = pk2(d2.z, d2.w);
            const unsigned int w6 = pk2(d3.x, d3.y), w7 = pk2(d3.z, d3.w);

            // Cross-half exchange between lane l and l^16:
            //   lo sends features 8..15 (w4..w7), hi sends 16..23 (w0..w3).
            const unsigned int s0 = hi ? w0 : w4;
            const unsigned int s1 = hi ? w1 : w5;
            const unsigned int s2 = hi ? w2 : w6;
            const unsigned int s3 = hi ? w3 : w7;
            const unsigned int r0 = (unsigned int)__builtin_amdgcn_ds_swizzle((int)s0, SWAPX16);
            const unsigned int r1 = (unsigned int)__builtin_amdgcn_ds_swizzle((int)s1, SWAPX16);
            const unsigned int r2 = (unsigned int)__builtin_amdgcn_ds_swizzle((int)s2, SWAPX16);
            const unsigned int r3 = (unsigned int)__builtin_amdgcn_ds_swizzle((int)s3, SWAPX16);

            // B operand (col N=lane&15): K = 0..15 (lo) / 16..31 (hi) = own w.
            V16H b;
            b.u[0] = w0; b.u[1] = w1; b.u[2] = w2; b.u[3] = w3;
            b.u[4] = w4; b.u[5] = w5; b.u[6] = w6; b.u[7] = w7;

            // A operand (row M=lane&15):
            //   K = 0..15 (lo) / 16..31 (hi) = own w (same order as B)
            V16H a;
            a.u[0] = w0; a.u[1] = w1;
            a.u[2] = w2; a.u[3] = w3;
            a.u[4] = w4; a.u[5] = w5;
            a.u[6] = w6; a.u[7] = w7;

            if (k == 0) {
                acc0 = __builtin_amdgcn_wmma_f32_16x16x32_f16(
                    false, a.v, false, b.v, (short)0, acc0, false, false);
            } else {
                acc1 = __builtin_amdgcn_wmma_f32_16x16x32_f16(
                    false, a.v, false, b.v, (short)0, acc1, false, false);
            }
        }
    }

    // Diagonal of the 16x16 f32 C/D tile:
    //   m in 0..7  -> lane m,      VGPR m
    //   m in 8..15 -> lane m + 16, VGPR m - 8
    int m = -1;
    if (lane < 8)        m = lane;
    else if (lane >= 24) m = lane - 16;
    if (m >= 0) {
        const int vg = m & 7;
        const float dmin = fminf(pick8(acc0, vg), pick8(acc1, vg));
        ws[sbase + m] = dmin;
    }
}

__global__ __launch_bounds__(1024)
void mcl_reduce_kernel(const float* __restrict__ ws, float* __restrict__ out) {
    __shared__ float sm[1024];
    const int t = threadIdx.x;
    float acc = 0.0f;
    for (int i = t; i < BATCH; i += 1024) acc += ws[i];   // fixed order: deterministic
    sm[t] = acc;
    __syncthreads();
    for (int off = 512; off > 0; off >>= 1) {
        if (t < off) sm[t] += sm[t + off];
        __syncthreads();
    }
    if (t == 0) out[0] = sm[0] * (1.0f / (float)BATCH);
}

extern "C" void kernel_launch(void* const* d_in, const int* in_sizes, int n_in,
                              void* d_out, int out_size, void* d_ws, size_t ws_size,
                              hipStream_t stream) {
    const float* x       = (const float*)d_in[0];
    const int*   labels  = (const int*)d_in[1];    // jax int64 -> int32 on device
    const float* centers = (const float*)d_in[2];
    float* out = (float*)d_out;
    float* ws  = (float*)d_ws;                     // needs >= BATCH*4 = 128 KB

    mcl_dist_kernel<<<BATCH / 128, 256, 0, stream>>>(x, labels, centers, ws);
    mcl_reduce_kernel<<<1, 1024, 0, stream>>>(ws, out);
}
